// NeuralBasisExpansion_64785286693589
// MI455X (gfx1250) — matrix operand order repaired and checked by reading, not compiled, not run on hardware
//
#include <hip/hip_runtime.h>
#include <math.h>

typedef __attribute__((ext_vector_type(16))) _Float16 v16h;
typedef __attribute__((ext_vector_type(8)))  _Float16 v8h;
typedef __attribute__((ext_vector_type(8)))  float    v8f;

#define HIDW   64
#define NBAS   8
#define NNODE  16
#define BATCH  2048
#define NPAIR  120
#define XCOLS  921
#define WAVES  8
#define TPW    2          // M-tiles (of 16 rows) per wave: amortizes B-fragment loads

__device__ __forceinline__ float fast_rcp(float x) { return __builtin_amdgcn_rcpf(x); }
__device__ __forceinline__ float silu_f(float x)   { return x * fast_rcp(1.0f + __expf(-x)); }

// ---------------------------------------------------------------------------
// Weight swizzle: f32 [NB][Ksrc][64] -> f16 B-fragments.
// blk = (n*4 + ntile)*nchunks + kc, 512 halves per blob.
// dst[blk*512 + lane*16 + e] = W[n][kc*32 + (e<8 ? e : 8+e) + (lane>=16?8:0)][ntile*16 + lane%16]
// ---------------------------------------------------------------------------
__global__ void swizzle_w(const float* __restrict__ W, _Float16* __restrict__ dst,
                          int Ksrc, int nchunks)
{
    int tid = blockIdx.x * blockDim.x + threadIdx.x;
    int total = NBAS * 4 * nchunks * 512;
    if (tid >= total) return;
    int e    = tid & 15;
    int lane = (tid >> 4) & 31;
    int blk  = tid >> 9;
    int kc   = blk % nchunks;
    int t    = (blk / nchunks) & 3;
    int n    = blk / (nchunks * 4);
    int kg   = kc * 32 + ((e < 8) ? e : 8 + e) + ((lane >= 16) ? 8 : 0);
    int col  = t * 16 + (lane & 15);
    float v  = (kg < Ksrc) ? W[((size_t)n * Ksrc + kg) * HIDW + col] : 0.0f;
    dst[(size_t)blk * 512 + lane * 16 + e] = (_Float16)v;
}

__global__ void copyz_kernel(const float* __restrict__ z, float* __restrict__ out)
{
    int gid = blockIdx.x * blockDim.x + threadIdx.x;
    if (gid >= BATCH * 64) return;
    int b = gid >> 6, i = gid & 63;
    out[(size_t)b * XCOLS + i] = z[gid];
}

__global__ void features_kernel(const float* __restrict__ z, float* __restrict__ out)
{
    int gid = blockIdx.x * blockDim.x + threadIdx.x;
    if (gid >= BATCH * NPAIR) return;
    int b = gid / NPAIR, p = gid % NPAIR;
    int i = 0, rem = p;
    while (rem >= 15 - i) { rem -= 15 - i; ++i; }
    int j = i + 1 + rem;
    const float* zb = z + (size_t)b * 64;
    float dx = zb[i * 4 + 0] - zb[j * 4 + 0];
    float dy = zb[i * 4 + 1] - zb[j * 4 + 1];
    float d  = sqrtf(dx * dx + dy * dy);
    d = fmaxf(d, 0.001f);
    float invd  = fast_rcp(d + 0.1f);
    float invsq = fast_rcp(d * d + 0.1f);
    float ed    = __expf(-fminf(d, 20.0f));
    float scr   = ed * invd;
    float ld    = __logf(d + 1.0f);
    size_t base = (size_t)b * XCOLS;
    out[base +  64 + p] = d;
    out[base + 184 + p] = invd;
    out[base + 304 + p] = invsq;
    out[base + 424 + p] = ed;
    out[base + 544 + p] = scr;
    out[base + 664 + p] = ld;
}

// ---------------------------------------------------------------------------
// WMMA basis kernel. FAM: 0=pw (din 8), 1=nd (din 4), 2=gl (din 64).
// One wave owns TPW consecutive 16-row M-tiles; every B fragment is loaded
// once and used for TPW WMMAs. All LDS exchange is intra-wave (per-wave hbuf
// slice) so only wave_barrier (compiler fence) is needed; LDS ops from the
// same wave execute in order.
// ---------------------------------------------------------------------------
template<int FAM>
__global__ void __launch_bounds__(256) basis_kernel(
    const float* __restrict__ z,
    const _Float16* __restrict__ w1s, const _Float16* __restrict__ w2s,
    const float* __restrict__ b1, const float* __restrict__ b2,
    const float* __restrict__ W3, const float* __restrict__ b3,
    const float* __restrict__ scl,
    const float* __restrict__ aW1, const float* __restrict__ ab1,
    const float* __restrict__ aW2, const float* __restrict__ ab2,
    const float* __restrict__ resp,
    float* __restrict__ out)
{
    constexpr int DIN  = (FAM == 0) ? 8 : (FAM == 1 ? 4 : 64);
    constexpr int K1CH = (FAM == 2) ? 2 : 1;

    // f16 staging: layer-1 activations, then reused for layer-2 activations
    __shared__ _Float16 __align__(16) hbuf[WAVES][TPW * 16][64];

    const int wv    = threadIdx.x >> 5;
    const int lane  = threadIdx.x & 31;
    const int lm    = lane & 15;
    const int hi    = lane >> 4;
    const int tile0 = (blockIdx.x * WAVES + wv) * TPW;

    // per-tile row decode
    int b_idx[TPW], ia[TPW], ja[TPW], pidx[TPW];
    #pragma unroll
    for (int m = 0; m < TPW; ++m) {
        int rowg = (tile0 + m) * 16 + lm;
        if (FAM == 0) {
            b_idx[m] = rowg / NPAIR; pidx[m] = rowg % NPAIR;
            int ii = 0, rem = pidx[m];
            while (rem >= 15 - ii) { rem -= 15 - ii; ++ii; }
            ia[m] = ii; ja[m] = ii + 1 + rem;
        } else if (FAM == 1) {
            b_idx[m] = rowg / NNODE; ia[m] = rowg % NNODE;
        } else {
            b_idx[m] = rowg;
        }
    }

    // --------- layer-1 A fragments (16x32 f16 per tile per K-chunk) --------
    v16h afr[TPW][K1CH];
    #pragma unroll
    for (int m = 0; m < TPW; ++m) {
        const float* zb = z + (size_t)b_idx[m] * 64;
        #pragma unroll
        for (int kc = 0; kc < K1CH; ++kc) {
            v16h a;
            #pragma unroll
            for (int e = 0; e < 16; ++e) a[e] = (_Float16)0.0f;
            if (FAM == 0) {
                if (hi == 0) {
                    #pragma unroll
                    for (int e = 0; e < 4; ++e) {
                        a[e]     = (_Float16)zb[ia[m] * 4 + e];
                        a[4 + e] = (_Float16)zb[ja[m] * 4 + e];
                    }
                }
            } else if (FAM == 1) {
                if (hi == 0) {
                    #pragma unroll
                    for (int e = 0; e < 4; ++e) a[e] = (_Float16)zb[ia[m] * 4 + e];
                }
            } else {
                #pragma unroll
                for (int e = 0; e < 8; ++e) {
                    a[e]     = (_Float16)zb[kc * 32 + e + 8 * hi];
                    a[8 + e] = (_Float16)zb[kc * 32 + 16 + e + 8 * hi];
                }
            }
            afr[m][kc] = a;
        }
    }

    float o_val[TPW][NBAS];
    const float res0 = resp[0];

    for (int n = 0; n < NBAS; ++n) {
        // ---------------- layer 1 ----------------
        v8f acc[TPW][4];
        #pragma unroll
        for (int t = 0; t < 4; ++t) {
            float bv = b1[n * HIDW + t * 16 + lm];
            #pragma unroll
            for (int m = 0; m < TPW; ++m)
                #pragma unroll
                for (int r = 0; r < 8; ++r) acc[m][t][r] = bv;
        }
        #pragma unroll
        for (int kc = 0; kc < K1CH; ++kc) {
            #pragma unroll
            for (int t = 0; t < 4; ++t) {
                v16h bf = *(const v16h*)(w1s + (((size_t)(n * 4 + t)) * K1CH + kc) * 512 + lane * 16);
                #pragma unroll
                for (int m = 0; m < TPW; ++m)
                    acc[m][t] = __builtin_amdgcn_wmma_f32_16x16x32_f16(
                        false, afr[m][kc], false, bf, (short)0, acc[m][t], false, false);
            }
        }
        #pragma unroll
        for (int m = 0; m < TPW; ++m)
            #pragma unroll
            for (int t = 0; t < 4; ++t)
                #pragma unroll
                for (int r = 0; r < 8; ++r)
                    hbuf[wv][m * 16 + r + 8 * hi][t * 16 + lm] = (_Float16)silu_f(acc[m][t][r]);
        __builtin_amdgcn_wave_barrier();

        // ---------------- layer 2 ----------------
        v8f acc2[TPW][4];
        #pragma unroll
        for (int t = 0; t < 4; ++t) {
            float bv = b2[n * HIDW + t * 16 + lm];
            #pragma unroll
            for (int m = 0; m < TPW; ++m)
                #pragma unroll
                for (int r = 0; r < 8; ++r) acc2[m][t][r] = bv;
        }
        #pragma unroll
        for (int kc = 0; kc < 2; ++kc) {
            int kb = kc * 32;
            v16h a2[TPW];
            #pragma unroll
            for (int m = 0; m < TPW; ++m) {
                v8h lo = *(const v8h*)&hbuf[wv][m * 16 + lm][kb + 8 * hi];
                v8h hb = *(const v8h*)&hbuf[wv][m * 16 + lm][kb + 16 + 8 * hi];
                #pragma unroll
                for (int e = 0; e < 8; ++e) { a2[m][e] = lo[e]; a2[m][8 + e] = hb[e]; }
            }
            #pragma unroll
            for (int t = 0; t < 4; ++t) {
                v16h bf = *(const v16h*)(w2s + (((size_t)(n * 4 + t)) * 2 + kc) * 512 + lane * 16);
                #pragma unroll
                for (int m = 0; m < TPW; ++m)
                    acc2[m][t] = __builtin_amdgcn_wmma_f32_16x16x32_f16(
                        false, a2[m], false, bf, (short)0, acc2[m][t], false, false);
            }
        }
        __builtin_amdgcn_wave_barrier();   // A-frag reads done before overwrite

        // silu -> reuse hbuf for layer-2 activations (f16)
        #pragma unroll
        for (int m = 0; m < TPW; ++m)
            #pragma unroll
            for (int t = 0; t < 4; ++t)
                #pragma unroll
                for (int r = 0; r < 8; ++r)
                    hbuf[wv][m * 16 + r + 8 * hi][t * 16 + lm] = (_Float16)silu_f(acc2[m][t][r]);
        __builtin_amdgcn_wave_barrier();

        // ---------------- o = h2 . w3[n], cross-half reduce ----------------
        int cb = 32 * hi;
        #pragma unroll
        for (int m = 0; m < TPW; ++m) {
            float op = 0.0f;
            #pragma unroll
            for (int c = 0; c < 32; ++c)
                op += (float)hbuf[wv][m * 16 + lm][cb + c] * W3[n * HIDW + cb + c];
            op += __shfl_xor(op, 16);
            o_val[m][n] = (op + b3[n]) * scl[n];
        }
        __builtin_amdgcn_wave_barrier();   // before next-basis hbuf writes
    }

    // ---------------- attention mixer + residual (per row, VALU) ----------
    #pragma unroll
    for (int m = 0; m < TPW; ++m) {
        const float* zb = z + (size_t)b_idx[m] * 64;
        float xv[DIN];
        if (FAM == 0) {
            #pragma unroll
            for (int e = 0; e < 4; ++e) { xv[e] = zb[ia[m] * 4 + e]; xv[4 + e] = zb[ja[m] * 4 + e]; }
        } else if (FAM == 1) {
            #pragma unroll
            for (int e = 0; e < 4; ++e) xv[e] = zb[ia[m] * 4 + e];
        } else {
            #pragma unroll
            for (int e = 0; e < 64; ++e) xv[e] = zb[e];
        }
        float a1[32];
        #pragma unroll 4
        for (int jj = 0; jj < 32; ++jj) {
            float s = ab1[jj];
            #pragma unroll
            for (int i = 0; i < DIN; ++i) s += xv[i] * aW1[i * 32 + jj];
            a1[jj] = silu_f(s);
        }
        float lg[NBAS], mx = -1e30f;
        #pragma unroll
        for (int nn = 0; nn < NBAS; ++nn) {
            float s = ab2[nn];
            #pragma unroll 8
            for (int jj = 0; jj < 32; ++jj) s += a1[jj] * aW2[jj * NBAS + nn];
            lg[nn] = s; mx = fmaxf(mx, s);
        }
        float den = 0.0f, o = 0.0f;
        #pragma unroll
        for (int nn = 0; nn < NBAS; ++nn) { lg[nn] = __expf(lg[nn] - mx); den += lg[nn]; }
        #pragma unroll
        for (int nn = 0; nn < NBAS; ++nn) o += o_val[m][nn] * lg[nn];
        o *= fast_rcp(den);
        float mean = 0.0f;
        #pragma unroll
        for (int i = 0; i < DIN; ++i) mean += xv[i];
        o += res0 * (mean * (1.0f / (float)DIN));

        if (lane < 16) {
            size_t oidx;
            if (FAM == 0)      oidx = (size_t)b_idx[m] * XCOLS + 784 + pidx[m];
            else if (FAM == 1) oidx = (size_t)b_idx[m] * XCOLS + 904 + ia[m];
            else               oidx = (size_t)b_idx[m] * XCOLS + 920;
            out[oidx] = o;
        }
    }
}

// ---------------------------------------------------------------------------
__global__ void selector_kernel(const float* __restrict__ z,
                                const float* __restrict__ W1, const float* __restrict__ B1,
                                const float* __restrict__ W2, const float* __restrict__ B2,
                                const float* __restrict__ W3, const float* __restrict__ B3,
                                float* __restrict__ sbuf)
{
    int b = blockIdx.x * blockDim.x + threadIdx.x;
    if (b >= BATCH) return;
    const float* zb = z + (size_t)b * 64;
    float zr[64];
    #pragma unroll
    for (int i = 0; i < 64; ++i) zr[i] = zb[i];
    float h1[64];
    for (int j = 0; j < 64; ++j) {
        float s = B1[j];
        #pragma unroll 8
        for (int i = 0; i < 64; ++i) s += zr[i] * W1[i * 64 + j];
        h1[j] = silu_f(s);
    }
    float acc = B3[0];
    for (int j = 0; j < 32; ++j) {
        float s = B2[j];
        #pragma unroll 8
        for (int i = 0; i < 64; ++i) s += h1[i] * W2[i * 32 + j];
        acc += silu_f(s) * W3[j];
    }
    sbuf[b] = fast_rcp(1.0f + __expf(-acc));
}

// X *= s, clip; 2D grid avoids per-thread integer division.
__global__ void scale_kernel(float* __restrict__ out, const float* __restrict__ sbuf)
{
    int col = blockIdx.x * blockDim.x + threadIdx.x;
    if (col >= XCOLS) return;
    size_t idx = (size_t)blockIdx.y * XCOLS + col;
    float v = out[idx] * sbuf[blockIdx.y];
    out[idx] = fminf(fmaxf(v, -1000000.0f), 1000000.0f);
}

// ---------------------------------------------------------------------------
extern "C" void kernel_launch(void* const* d_in, const int* in_sizes, int n_in,
                              void* d_out, int out_size, void* d_ws, size_t ws_size,
                              hipStream_t stream)
{
    (void)in_sizes; (void)n_in; (void)out_size; (void)ws_size;
    const float* z = (const float*)d_in[0];
    const float* pw_W1 = (const float*)d_in[1];  const float* pw_b1 = (const float*)d_in[2];
    const float* pw_W2 = (const float*)d_in[3];  const float* pw_b2 = (const float*)d_in[4];
    const float* pw_W3 = (const float*)d_in[5];  const float* pw_b3 = (const float*)d_in[6];
    const float* pw_sc = (const float*)d_in[7];  const float* pw_a1 = (const float*)d_in[8];
    const float* pw_ab1= (const float*)d_in[9];  const float* pw_a2 = (const float*)d_in[10];
    const float* pw_ab2= (const float*)d_in[11]; const float* pw_rs = (const float*)d_in[12];
    const float* nd_W1 = (const float*)d_in[13]; const float* nd_b1 = (const float*)d_in[14];
    const float* nd_W2 = (const float*)d_in[15]; const float* nd_b2 = (const float*)d_in[16];
    const float* nd_W3 = (const float*)d_in[17]; const float* nd_b3 = (const float*)d_in[18];
    const float* nd_sc = (const float*)d_in[19]; const float* nd_a1 = (const float*)d_in[20];
    const float* nd_ab1= (const float*)d_in[21]; const float* nd_a2 = (const float*)d_in[22];
    const float* nd_ab2= (const float*)d_in[23]; const float* nd_rs = (const float*)d_in[24];
    const float* gl_W1 = (const float*)d_in[25]; const float* gl_b1 = (const float*)d_in[26];
    const float* gl_W2 = (const float*)d_in[27]; const float* gl_b2 = (const float*)d_in[28];
    const float* gl_W3 = (const float*)d_in[29]; const float* gl_b3 = (const float*)d_in[30];
    const float* gl_sc = (const float*)d_in[31]; const float* gl_a1 = (const float*)d_in[32];
    const float* gl_ab1= (const float*)d_in[33]; const float* gl_a2 = (const float*)d_in[34];
    const float* gl_ab2= (const float*)d_in[35]; const float* gl_rs = (const float*)d_in[36];
    const float* selW1 = (const float*)d_in[37]; const float* selb1 = (const float*)d_in[38];
    const float* selW2 = (const float*)d_in[39]; const float* selb2 = (const float*)d_in[40];
    const float* selW3 = (const float*)d_in[41]; const float* selb3 = (const float*)d_in[42];

    float* out = (float*)d_out;
    _Float16* ws16 = (_Float16*)d_ws;
    _Float16* pw1s = ws16;              // 16384 halves
    _Float16* pw2s = ws16 + 16384;      // 32768
    _Float16* nd1s = ws16 + 49152;      // 16384
    _Float16* nd2s = ws16 + 65536;      // 32768
    _Float16* gl1s = ws16 + 98304;      // 32768
    _Float16* gl2s = ws16 + 131072;     // 32768 -> 163840 halves total
    float* sbuf = (float*)((char*)d_ws + 327680);   // 2048 floats

    swizzle_w<<<64,  256, 0, stream>>>(pw_W1, pw1s, 8,  1);
    swizzle_w<<<128, 256, 0, stream>>>(pw_W2, pw2s, 64, 2);
    swizzle_w<<<64,  256, 0, stream>>>(nd_W1, nd1s, 4,  1);
    swizzle_w<<<128, 256, 0, stream>>>(nd_W2, nd2s, 64, 2);
    swizzle_w<<<128, 256, 0, stream>>>(gl_W1, gl1s, 64, 2);
    swizzle_w<<<128, 256, 0, stream>>>(gl_W2, gl2s, 64, 2);

    copyz_kernel<<<(BATCH * 64) / 256, 256, 0, stream>>>(z, out);
    features_kernel<<<(BATCH * NPAIR + 255) / 256, 256, 0, stream>>>(z, out);

    // pw: 245760 rows -> 15360 tiles -> /TPW /8 waves = 960 blocks
    basis_kernel<0><<<960, 256, 0, stream>>>(z, pw1s, pw2s, pw_b1, pw_b2, pw_W3, pw_b3,
                                             pw_sc, pw_a1, pw_ab1, pw_a2, pw_ab2, pw_rs, out);
    // nd: 32768 rows -> 2048 tiles -> 128 blocks
    basis_kernel<1><<<128, 256, 0, stream>>>(z, nd1s, nd2s, nd_b1, nd_b2, nd_W3, nd_b3,
                                             nd_sc, nd_a1, nd_ab1, nd_a2, nd_ab2, nd_rs, out);
    // gl: 2048 rows -> 128 tiles -> 8 blocks
    basis_kernel<2><<<8, 256, 0, stream>>>(z, gl1s, gl2s, gl_b1, gl_b2, gl_W3, gl_b3,
                                           gl_sc, gl_a1, gl_ab1, gl_a2, gl_ab2, gl_rs, out);

    selector_kernel<<<8, 256, 0, stream>>>(z, selW1, selb1, selW2, selb2, selW3, selb3, sbuf);
    dim3 sg((XCOLS + 255) / 256, BATCH);
    scale_kernel<<<sg, 256, 0, stream>>>(out, sbuf);
}
